// VICRegLLoss_24833500905723
// MI455X (gfx1250) — matrix operand order repaired and checked by reading, not compiled, not run on hardware
//
#include <hip/hip_runtime.h>
#include <math.h>

typedef __attribute__((ext_vector_type(2)))  float  v2f;
typedef __attribute__((ext_vector_type(8)))  float  v8f;
typedef __attribute__((ext_vector_type(16))) __bf16 v16bf;

#define B_    128
#define D_    8192
#define NLOC  49
#define NPAD  64
#define M20   20
#define M4    4
#define PITCH 130   // k-pitch for LDS panels: 130 mod 64 = 2 -> conflict-free column access

// workspace layout (float index space)
#define ACC_OFF     0              // 16 floats of accumulators
#define MEANA_OFF   16
#define MEANB_OFF   (16 + D_)
#define VARA_OFF    (16 + 2*D_)
#define VARB_OFF    (16 + 3*D_)
#define INT_OFF     (16 + 4*D_)    // int region starts here
// int region offsets (in ints)
#define NNFA_OFF    0              // 128*49
#define SELFA_OFF   6272           // 128*20
#define NNFB_OFF    8832
#define SELFB_OFF   15104
#define NNGA_OFF    17664
#define SELGA_OFF   23936
#define NNGB_OFF    26496
#define SELGB_OFF   32768          // 128*4

// ---------------- init accumulators ----------------
__global__ void k_init(float* ws) {
  if (threadIdx.x < 16) ws[ACC_OFF + threadIdx.x] = 0.f;
}

// ---------------- global invariance MSE ----------------
__global__ void k_global_inv(const float* a, const float* b, float* ws) {
  int idx = blockIdx.x * blockDim.x + threadIdx.x;
  int stride = gridDim.x * blockDim.x;
  float s = 0.f;
  for (int i = idx; i < B_ * D_; i += stride) {
    float d = a[i] - b[i];
    s += d * d;
  }
  for (int o = 16; o; o >>= 1) s += __shfl_xor(s, o, 32);
  if ((threadIdx.x & 31) == 0) atomicAdd(&ws[ACC_OFF + 0], s);
}

// ---------------- per-dim mean / var (ddof=1), variance loss, diag^2 ----------------
__global__ void k_meanvar(const float* za, const float* zb, float* ws) {
  const float* X = (blockIdx.y == 0) ? za : zb;
  float* mean = ws + ((blockIdx.y == 0) ? MEANA_OFF : MEANB_OFF);
  float* var  = ws + ((blockIdx.y == 0) ? VARA_OFF  : VARB_OFF);
  int d = blockIdx.x * blockDim.x + threadIdx.x;
  if (d >= D_) return;
  float s = 0.f, ss = 0.f;
  for (int k = 0; k < B_; ++k) {
    float v = X[(size_t)k * D_ + d];
    s += v; ss += v * v;
  }
  float m  = s / (float)B_;
  float vr = (ss - s * m) / (float)(B_ - 1);
  mean[d] = m;
  var[d]  = vr;
  float sd = sqrtf(vr + 1e-4f);
  float r  = fmaxf(0.f, 1.f - sd);
  atomicAdd(&ws[ACC_OFF + 1 + blockIdx.y], r);        // variance-loss sum
  atomicAdd(&ws[ACC_OFF + 3 + blockIdx.y], vr * vr);  // sum of diag(cov)^2
}

// ---------------- covariance Gram via exact f32 WMMA, LDS-staged panels ----------------
// G = Xc^T * Xc (unnormalized). Block = 8 waves; all 8 tiles share one I panel
// (512 J-tiles per I row is divisible by 8, so I is uniform per block).
// Panels stored k-contiguous: panel[c*PITCH + k], centered on store.
// A(16x4) f32 layout: lanes0-15 -> K={0,1}, lanes16-31 -> K={2,3}, row M = lane&15.
// B(4x16) f32 layout: VGPR0 K = (lane<16?0:2), VGPR1 K = (lane<16?1:3), col N = lane&15.
__global__ void __launch_bounds__(256) k_cov_wmma(const float* za, const float* zb, float* ws) {
  __shared__ float aP[16 * PITCH];
  __shared__ float bP[8][16 * PITCH];
  const float* X    = (blockIdx.y == 0) ? za : zb;
  const float* mean = ws + ((blockIdx.y == 0) ? MEANA_OFF : MEANB_OFF);
  int wave = threadIdx.x >> 5;
  int lane = threadIdx.x & 31;
  int tile0 = blockIdx.x * 8;
  int I     = tile0 >> 9;        // 512 tiles per dimension (8192/16)
  int Jbase = tile0 & 511;

  // stage A panel: columns I*16..I*16+15, k = 0..127 (coalesced reads, scattered LDS writes)
  for (int t = threadIdx.x; t < 16 * B_; t += 256) {
    int k = t >> 4, c = t & 15;
    int col = I * 16 + c;
    aP[c * PITCH + k] = X[(size_t)k * D_ + col] - mean[col];
  }
  // stage the 8 B panels
  for (int p = 0; p < 8; ++p) {
    int colbase = (Jbase + p) * 16;
    for (int t = threadIdx.x; t < 16 * B_; t += 256) {
      int k = t >> 4, c = t & 15;
      int col = colbase + c;
      bP[p][c * PITCH + k] = X[(size_t)k * D_ + col] - mean[col];
    }
  }
  __syncthreads();

  int m   = lane & 15;
  int hi  = lane >> 4;
  int kbA = hi * 2;               // A lane K base: 0 or 2
  int kbB0 = hi ? 2 : 0;          // B vgpr0 K
  const float* aL = &aP[m * PITCH];
  const float* bL = &bP[wave][m * PITCH];
  v8f c = {};
#pragma unroll
  for (int k0 = 0; k0 < B_; k0 += 4) {
    v2f a, b;
    a.x = aL[k0 + kbA];           // consecutive k -> single ds_load_b64
    a.y = aL[k0 + kbA + 1];
    b.x = bL[k0 + kbB0];
    b.y = bL[k0 + kbB0 + 1];
    c = __builtin_amdgcn_wmma_f32_16x16x4_f32(false, a, false, b, (short)0, c, false, false);
  }
  float s = 0.f;
#pragma unroll
  for (int v = 0; v < 8; ++v) s += c[v] * c[v];
  for (int o = 16; o; o >>= 1) s += __shfl_xor(s, o, 32);
  if (lane == 0) atomicAdd(&ws[ACC_OFF + 5 + blockIdx.y], s);  // sum_{ij} G_ij^2
}

// ---------------- per-batch feature Gram (bf16 hi/lo WMMA) + NN selection -------------
// A(16x32 bf16): lane<16 holds row m, K in {0..7, 16..23}; lane>=16: K in {8..15, 24..31}.
// B(32x16 bf16): lane holds column n=lane&15, same K set as A: lane<16 K in {0..7, 16..23}; lane>=16: K in {8..15, 24..31}.
__global__ void __launch_bounds__(256) k_feat_match(const float* za, const float* zb, float* ws) {
  __shared__ float S[NPAD * NPAD];
  __shared__ float na[NPAD], nb[NPAD];
  __shared__ float nnval[NLOC];
  __shared__ int   nnidx[NLOC];
  __shared__ unsigned char used[NLOC];
  int b = blockIdx.x;
  int wave = threadIdx.x >> 5;
  int lane = threadIdx.x & 31;
  const float* A  = za + (size_t)b * NLOC * D_;
  const float* Bp = zb + (size_t)b * NLOC * D_;
  int* ip = (int*)(ws + INT_OFF);
  int m  = lane & 15;
  int kh = lane >> 4;

  for (int r = 0; r < 2; ++r) {
    int tile = wave * 2 + r;
    int ti = tile >> 2, tj = tile & 3;
    int rowA = ti * 16 + m;
    int colB = tj * 16 + m;
    v8f c = {};
    for (int k0 = 0; k0 < D_; k0 += 32) {
      v16bf ah, al, bh, bl;
#pragma unroll
      for (int e = 0; e < 16; ++e) {
        int K = ((e >> 3) << 4) + (kh << 3) + (e & 7);
        float x = (rowA < NLOC) ? A[(size_t)rowA * D_ + k0 + K] : 0.f;
        __bf16 h = (__bf16)x;
        ah[e] = h;
        al[e] = (__bf16)(x - (float)h);
      }
#pragma unroll
      for (int e = 0; e < 16; ++e) {
        int K = ((e >> 3) << 4) + (kh << 3) + (e & 7);
        float x = (colB < NLOC) ? Bp[(size_t)colB * D_ + k0 + K] : 0.f;
        __bf16 h = (__bf16)x;
        bh[e] = h;
        bl[e] = (__bf16)(x - (float)h);
      }
      c = __builtin_amdgcn_wmma_f32_16x16x32_bf16(false, ah, false, bh, (short)0, c, false, false);
      c = __builtin_amdgcn_wmma_f32_16x16x32_bf16(false, ah, false, bl, (short)0, c, false, false);
      c = __builtin_amdgcn_wmma_f32_16x16x32_bf16(false, al, false, bh, (short)0, c, false, false);
    }
#pragma unroll
    for (int v = 0; v < 8; ++v)
      S[(ti * 16 + v + kh * 8) * NPAD + tj * 16 + m] = c[v];
  }
  // squared norms
  for (int r = wave; r < NLOC; r += 8) {
    float s = 0.f;
    for (int k = lane; k < D_; k += 32) { float v = A[(size_t)r * D_ + k]; s += v * v; }
    for (int o = 16; o; o >>= 1) s += __shfl_xor(s, o, 32);
    if (lane == 0) na[r] = s;
  }
  for (int r = wave; r < NLOC; r += 8) {
    float s = 0.f;
    for (int k = lane; k < D_; k += 32) { float v = Bp[(size_t)r * D_ + k]; s += v * v; }
    for (int o = 16; o; o >>= 1) s += __shfl_xor(s, o, 32);
    if (lane == 0) nb[r] = s;
  }
  __syncthreads();
  // a -> b argmin
  if (threadIdx.x < NLOC) {
    int i = threadIdx.x;
    float best = 3.4e38f; int bj = 0;
    for (int j = 0; j < NLOC; ++j) {
      float d2 = na[i] + nb[j] - 2.f * S[i * NPAD + j];
      if (d2 < best) { best = d2; bj = j; }
    }
    nnidx[i] = bj; nnval[i] = best;
    ip[NNFA_OFF + b * NLOC + i] = bj;
  }
  __syncthreads();
  if (threadIdx.x == 0) {
    for (int i = 0; i < NLOC; ++i) used[i] = 0;
    for (int t = 0; t < M20; ++t) {
      float best = 3.4e38f; int bi = 0;
      for (int i = 0; i < NLOC; ++i)
        if (!used[i] && nnval[i] < best) { best = nnval[i]; bi = i; }
      used[bi] = 1;
      ip[SELFA_OFF + b * M20 + t] = bi;
    }
  }
  __syncthreads();
  // b -> a argmin (transpose of S)
  if (threadIdx.x < NLOC) {
    int i = threadIdx.x;
    float best = 3.4e38f; int bj = 0;
    for (int j = 0; j < NLOC; ++j) {
      float d2 = nb[i] + na[j] - 2.f * S[j * NPAD + i];
      if (d2 < best) { best = d2; bj = j; }
    }
    nnidx[i] = bj; nnval[i] = best;
    ip[NNFB_OFF + b * NLOC + i] = bj;
  }
  __syncthreads();
  if (threadIdx.x == 0) {
    for (int i = 0; i < NLOC; ++i) used[i] = 0;
    for (int t = 0; t < M20; ++t) {
      float best = 3.4e38f; int bi = 0;
      for (int i = 0; i < NLOC; ++i)
        if (!used[i] && nnval[i] < best) { best = nnval[i]; bi = i; }
      used[bi] = 1;
      ip[SELFB_OFF + b * M20 + t] = bi;
    }
  }
}

// ---------------- per-batch grid (spatial) NN selection ----------------
__global__ void k_grid_match(const float* ga, const float* gb, float* ws) {
  __shared__ float gax[NLOC], gay[NLOC], gbx[NLOC], gby[NLOC];
  __shared__ float nnval[NLOC];
  __shared__ unsigned char used[NLOC];
  int b = blockIdx.x;
  int* ip = (int*)(ws + INT_OFF);
  if (threadIdx.x < NLOC) {
    int i = threadIdx.x;
    gax[i] = ga[((size_t)b * NLOC + i) * 2 + 0];
    gay[i] = ga[((size_t)b * NLOC + i) * 2 + 1];
    gbx[i] = gb[((size_t)b * NLOC + i) * 2 + 0];
    gby[i] = gb[((size_t)b * NLOC + i) * 2 + 1];
  }
  __syncthreads();
  if (threadIdx.x < NLOC) {
    int i = threadIdx.x;
    float best = 3.4e38f; int bj = 0;
    for (int j = 0; j < NLOC; ++j) {
      float dx = gax[i] - gbx[j], dy = gay[i] - gby[j];
      float d2 = dx * dx + dy * dy;
      if (d2 < best) { best = d2; bj = j; }
    }
    nnval[i] = best;
    ip[NNGA_OFF + b * NLOC + i] = bj;
  }
  __syncthreads();
  if (threadIdx.x == 0) {
    for (int i = 0; i < NLOC; ++i) used[i] = 0;
    for (int t = 0; t < M20; ++t) {
      float best = 3.4e38f; int bi = 0;
      for (int i = 0; i < NLOC; ++i)
        if (!used[i] && nnval[i] < best) { best = nnval[i]; bi = i; }
      used[bi] = 1;
      ip[SELGA_OFF + b * M20 + t] = bi;
    }
  }
  __syncthreads();
  if (threadIdx.x < NLOC) {
    int i = threadIdx.x;
    float best = 3.4e38f; int bj = 0;
    for (int j = 0; j < NLOC; ++j) {
      float dx = gbx[i] - gax[j], dy = gby[i] - gay[j];
      float d2 = dx * dx + dy * dy;
      if (d2 < best) { best = d2; bj = j; }
    }
    nnval[i] = best;
    ip[NNGB_OFF + b * NLOC + i] = bj;
  }
  __syncthreads();
  if (threadIdx.x == 0) {
    for (int i = 0; i < NLOC; ++i) used[i] = 0;
    for (int t = 0; t < M4; ++t) {
      float best = 3.4e38f; int bi = 0;
      for (int i = 0; i < NLOC; ++i)
        if (!used[i] && nnval[i] < best) { best = nnval[i]; bi = i; }
      used[bi] = 1;
      ip[SELGB_OFF + b * M4 + t] = bi;
    }
  }
}

// ---------------- gathered-feature MSE accumulation ----------------
__global__ void k_mse_gather(const float* inmap, const float* candmap,
                             const int* sel, const int* nnidx, int M, float* slot) {
  int b = blockIdx.x, t = blockIdx.y;
  int i = sel[b * M + t];
  int j = nnidx[b * NLOC + i];
  const float* pi = inmap   + ((size_t)b * NLOC + i) * D_;
  const float* pj = candmap + ((size_t)b * NLOC + j) * D_;
  float s = 0.f;
  for (int k = threadIdx.x; k < D_; k += blockDim.x) {
    float d = pi[k] - pj[k];
    s += d * d;
  }
  for (int o = 16; o; o >>= 1) s += __shfl_xor(s, o, 32);
  __shared__ float red[8];
  if ((threadIdx.x & 31) == 0) red[threadIdx.x >> 5] = s;
  __syncthreads();
  if (threadIdx.x == 0) {
    float tot = 0.f;
    for (int w = 0; w < (int)(blockDim.x >> 5); ++w) tot += red[w];
    atomicAdd(slot, tot);
  }
}

// ---------------- final combine ----------------
__global__ void k_finalize(const float* ws, float* out) {
  const float* acc = ws + ACC_OFF;
  float inv_g = acc[0] / (float)(B_ * D_);
  float varl  = 0.5f * (acc[1] + acc[2]) / (float)D_;
  float inv127 = 1.f / 127.f;
  float cov_a = (acc[5] * inv127 * inv127 - acc[3]) / (float)D_;
  float cov_b = (acc[6] * inv127 * inv127 - acc[4]) / (float)D_;
  float g = 25.f * inv_g + 25.f * varl + 1.f * (cov_a + cov_b);
  float nf20 = (float)B_ * (float)M20 * (float)D_;
  float nf4  = (float)B_ * (float)M4  * (float)D_;
  float inv_l = 0.5f * acc[7] / nf20 + 0.5f * acc[8] / nf20
              + 0.5f * acc[9] / nf20 + 0.5f * acc[10] / nf4;
  float l = 25.f * inv_l;
  out[0] = 0.25f * g + 0.75f * l;
}

extern "C" void kernel_launch(void* const* d_in, const int* in_sizes, int n_in,
                              void* d_out, int out_size, void* d_ws, size_t ws_size,
                              hipStream_t stream) {
  const float* z_a  = (const float*)d_in[0];
  const float* z_b  = (const float*)d_in[1];
  const float* za_l = (const float*)d_in[2];
  const float* zb_l = (const float*)d_in[3];
  const float* ga   = (const float*)d_in[4];
  const float* gb   = (const float*)d_in[5];
  float* out = (float*)d_out;
  float* ws  = (float*)d_ws;
  int*   ip  = (int*)(ws + INT_OFF);

  k_init<<<1, 32, 0, stream>>>(ws);
  k_global_inv<<<512, 256, 0, stream>>>(z_a, z_b, ws);
  dim3 gmv(D_ / 256, 2);
  k_meanvar<<<gmv, 256, 0, stream>>>(z_a, z_b, ws);
  dim3 gcov((512 * 512) / 8, 2);
  k_cov_wmma<<<gcov, 256, 0, stream>>>(z_a, z_b, ws);
  k_feat_match<<<B_, 256, 0, stream>>>(za_l, zb_l, ws);
  k_grid_match<<<B_, 64, 0, stream>>>(ga, gb, ws);
  dim3 g20(B_, M20), g4(B_, M4);
  k_mse_gather<<<g20, 256, 0, stream>>>(za_l, zb_l, ip + SELFA_OFF, ip + NNFA_OFF, M20, ws + ACC_OFF + 7);
  k_mse_gather<<<g20, 256, 0, stream>>>(zb_l, za_l, ip + SELFB_OFF, ip + NNFB_OFF, M20, ws + ACC_OFF + 8);
  k_mse_gather<<<g20, 256, 0, stream>>>(za_l, zb_l, ip + SELGA_OFF, ip + NNGA_OFF, M20, ws + ACC_OFF + 9);
  k_mse_gather<<<g4,  256, 0, stream>>>(zb_l, za_l, ip + SELGB_OFF, ip + NNGB_OFF, M4,  ws + ACC_OFF + 10);
  k_finalize<<<1, 1, 0, stream>>>(ws, out);
}
